// PolylineEncoder_37374805409816
// MI455X (gfx1250) — compile-verified, matrix operand order checked by reading, not run on hardware
//
#include <hip/hip_runtime.h>

#define N_NODE   20
#define HIDDEN   256
#define HALF_H   128
#define GROUP_MP 4
#define ROWS     80                 // GROUP_MP * N_NODE -> exactly 5 M-tiles
#define M_TILES  5
#define K_STEPS  8                  // HIDDEN / 32
#define NTHREADS 256                // 8 waves of 32
#define NGROUPS  4096               // 16*1024 / GROUP_MP
#define XSTRIDE  264                // padded LDS row stride (f16), bank-friendly
#define PGSTRIDE 136                // pooled-tile row stride (f16), bank-friendly

typedef _Float16 v4h  __attribute__((ext_vector_type(4)));
typedef _Float16 v8h  __attribute__((ext_vector_type(8)));
typedef _Float16 v16h __attribute__((ext_vector_type(16)));
typedef float    v8f  __attribute__((ext_vector_type(8)));

// ---- prep: convert W0/W1/W2 (f32) to one contiguous f16 buffer in d_ws ----
__global__ void cvt_weights(const float* __restrict__ W0,
                            const float* __restrict__ W1,
                            const float* __restrict__ W2,
                            _Float16* __restrict__ out) {
  int i = blockIdx.x * blockDim.x + threadIdx.x;
  if (i >= 3 * HALF_H * HIDDEN) return;
  int l = i / (HALF_H * HIDDEN);
  int r = i % (HALF_H * HIDDEN);
  const float* s = (l == 0) ? W0 : (l == 1) ? W1 : W2;
  out[i] = (_Float16)s[r];
}

__global__ __launch_bounds__(NTHREADS, 1)
void polyline_encoder(const float* __restrict__ x,
                      const int* __restrict__ invalid,
                      const _Float16* __restrict__ wf16,
                      const float* __restrict__ b0,
                      const float* __restrict__ b1,
                      const float* __restrict__ b2,
                      float* __restrict__ out) {
  __shared__ _Float16 sX[ROWS * XSTRIDE];           // activation tile
  __shared__ unsigned sPool[GROUP_MP * HALF_H];     // f32-as-uint max pool
  __shared__ _Float16 sPG[16 * PGSTRIDE];           // compact pooled tile (4 mp + pad)
  __shared__ int      sInv[ROWS];

  const int tid  = threadIdx.x;
  const int wave = tid >> 5;
  const int L    = tid & 31;
  const int lm   = L & 15;        // row/col within 16
  const int half = L >> 4;        // lane half selects K sub-block
  const int wc   = wave * 16;     // this wave's output-channel base

  // ---- persistent weight registers: B operand, lane layout mirrors A (N x K) ----
  v16h Wreg[3][K_STEPS];
#pragma unroll
  for (int l = 0; l < 3; ++l) {
#pragma unroll
    for (int k = 0; k < K_STEPS; ++k) {
      const _Float16* base =
          wf16 + ((l * HALF_H + wc + lm) * HIDDEN) + k * 32 + half * 8;
      v8h w0 = *(const v8h*)(base);
      v8h w1 = *(const v8h*)(base + 16);
      Wreg[l][k] = __builtin_shufflevector(w0, w1,
          0,1,2,3,4,5,6,7,8,9,10,11,12,13,14,15);
    }
  }
  float bias_r[3];
  bias_r[0] = b0[wc + lm];
  bias_r[1] = b1[wc + lm];
  bias_r[2] = b2[wc + lm];

  // zero pooled tile once; pad rows 4..15 stay zero forever
  for (int i = tid; i < 16 * PGSTRIDE; i += NTHREADS) sPG[i] = (_Float16)0.f;

  for (int gi = blockIdx.x; gi < NGROUPS; gi += gridDim.x) {
    const int gbase = gi * GROUP_MP;                // flat (sc,mp) base index

    if (tid < ROWS) sInv[tid] = invalid[gbase * N_NODE + tid];
    for (int i = tid; i < GROUP_MP * HALF_H; i += NTHREADS) sPool[i] = 0u;

    // load 80x256 f32 rows, convert to f16 into LDS
    const float4* xg = (const float4*)(x + (size_t)gbase * N_NODE * HIDDEN);
#pragma unroll
    for (int it = 0; it < 20; ++it) {
      int q = it * NTHREADS + tid;                  // quad index 0..5119
      float4 v = xg[q];
      int row = q >> 6;
      int c   = (q & 63) * 4;
      *(v4h*)&sX[row * XSTRIDE + c] =
          (v4h){(_Float16)v.x, (_Float16)v.y, (_Float16)v.z, (_Float16)v.w};
    }
    __syncthreads();

    // prefetch next group's x into L2/WGP$ while we compute (no counter cost)
    if (gi + (int)gridDim.x < NGROUPS) {
      const char* nx =
          (const char*)(x + (size_t)(gi + gridDim.x) * GROUP_MP * N_NODE * HIDDEN);
#pragma unroll
      for (int it = 0; it < 3; ++it) {
        int line = it * NTHREADS + tid;             // 128B cachelines, 640 total
        if (line < (ROWS * HIDDEN * 4) / 128)
          __builtin_prefetch(nx + line * 128, 0, 3);
      }
    }

#pragma unroll
    for (int l = 0; l < 3; ++l) {
      const int ksteps = (l == 0) ? K_STEPS : (K_STEPS / 2);

      // per-mp correction term: W_right @ pooled  (layers 1,2 only)
      float term_r[4] = {0.f, 0.f, 0.f, 0.f};
      if (l > 0) {
        v8f tacc = {};
#pragma unroll
        for (int k = 0; k < K_STEPS / 2; ++k) {
          const _Float16* ab = &sPG[lm * PGSTRIDE + k * 32 + half * 8];
          v8h a0 = *(const v8h*)(ab);
          v8h a1 = *(const v8h*)(ab + 16);
          v16h A = __builtin_shufflevector(a0, a1,
              0,1,2,3,4,5,6,7,8,9,10,11,12,13,14,15);
          tacc = __builtin_amdgcn_wmma_f32_16x16x32_f16(
              false, A, false, Wreg[l][4 + k], (short)0, tacc, false, false);
        }
        // term[g][col] lives in element g of lane 'col' (half 0); broadcast it
#pragma unroll
        for (int g = 0; g < GROUP_MP; ++g) term_r[g] = __shfl(tacc[g], lm, 32);
      }

      v8f acc[M_TILES];
#pragma unroll
      for (int t = 0; t < M_TILES; ++t) {
        v8f c = {};
#pragma unroll
        for (int k = 0; k < K_STEPS; ++k) {
          if (k < ksteps) {
            const _Float16* ab =
                &sX[(t * 16 + lm) * XSTRIDE + k * 32 + half * 8];
            v8h a0 = *(const v8h*)(ab);
            v8h a1 = *(const v8h*)(ab + 16);
            v16h A = __builtin_shufflevector(a0, a1,
                0,1,2,3,4,5,6,7,8,9,10,11,12,13,14,15);
            c = __builtin_amdgcn_wmma_f32_16x16x32_f16(
                false, A, false, Wreg[l][k], (short)0, c, false, false);
          }
        }
        acc[t] = c;
      }

      // term + bias + relu + masked max-pool (relu >= 0 so uint-max == f32-max)
      const int col = wc + lm;
#pragma unroll
      for (int t = 0; t < M_TILES; ++t) {
#pragma unroll
        for (int i = 0; i < 8; ++i) {
          const int gA = (t * 16 + i) / N_NODE;       // compile-time constants
          const int gB = (t * 16 + i + 8) / N_NODE;
          float term = half ? term_r[gB] : term_r[gA];
          float h = acc[t][i] + term + bias_r[l];
          h = h > 0.f ? h : 0.f;
          acc[t][i] = h;
          int row = t * 16 + i + half * 8;
          if (!sInv[row])
            atomicMax(&sPool[(row / N_NODE) * HALF_H + col],
                      __float_as_uint(h));
        }
      }
      __syncthreads();  // all sX/sPG reads + pool updates complete

      if (l < 2) {
        // next activations: h into left half of sX (zeroed at invalid rows)
#pragma unroll
        for (int t = 0; t < M_TILES; ++t) {
#pragma unroll
          for (int i = 0; i < 8; ++i) {
            int row = t * 16 + i + half * 8;
            sX[row * XSTRIDE + col] =
                sInv[row] ? (_Float16)0.f : (_Float16)acc[t][i];
          }
        }
        // compact pooled tile for next layer's correction term
        for (int idx = tid; idx < GROUP_MP * HALF_H; idx += NTHREADS) {
          int g = idx >> 7, c = idx & 127;
          sPG[g * PGSTRIDE + c] = (_Float16)__uint_as_float(sPool[idx]);
        }
      } else {
        // final output = layer-2 pool duplicated in both halves;
        // all-invalid case is 0 automatically (pool stayed at 0-init)
        for (int idx = tid; idx < GROUP_MP * HIDDEN; idx += NTHREADS) {
          int m = idx >> 8, c = idx & 255;
          out[(size_t)(gbase + m) * HIDDEN + c] =
              __uint_as_float(sPool[m * HALF_H + (c & 127)]);
        }
      }
      __syncthreads();  // writes visible / pool reads done
      for (int i = tid; i < GROUP_MP * HALF_H; i += NTHREADS) sPool[i] = 0u;
      __syncthreads();
    }
  }
}

extern "C" void kernel_launch(void* const* d_in, const int* in_sizes, int n_in,
                              void* d_out, int out_size, void* d_ws, size_t ws_size,
                              hipStream_t stream) {
  const float* x  = (const float*)d_in[0];
  const int*   iv = (const int*)d_in[1];
  const float* W0 = (const float*)d_in[2];
  const float* b0 = (const float*)d_in[3];
  const float* W1 = (const float*)d_in[4];
  const float* b1 = (const float*)d_in[5];
  const float* W2 = (const float*)d_in[6];
  const float* b2 = (const float*)d_in[7];
  _Float16* wf16  = (_Float16*)d_ws;   // 3*128*256*2 = 192 KiB of scratch

  hipLaunchKernelGGL(cvt_weights, dim3(384), dim3(256), 0, stream,
                     W0, W1, W2, wf16);
  hipLaunchKernelGGL(polyline_encoder, dim3(1024), dim3(NTHREADS), 0, stream,
                     x, iv, wf16, b0, b1, b2, (float*)d_out);
}
